// HindiCausalSelfAttention_13348758356011
// MI455X (gfx1250) — matrix operand order repaired and checked by reading, not compiled, not run on hardware
//
#include <hip/hip_runtime.h>
#include <hip/hip_bf16.h>

// ---------------------------------------------------------------------------
// Causal MHA for MI455X (gfx1250): bf16 WMMA pipeline, f32 accumulate.
// B=2, S=2048, D=1024, H=16, hd=64
// GEMMs: double-buffered LDS tiles staged with GLOBAL_LOAD_ASYNC_TO_LDS_B128
// (ASYNCcnt), overlapping tensor DMA with v_wmma issue.
// Attention: register-only flash attention, 32 queries per wave, transposed
// scores so P converts to the next WMMA's A-operand with per-lane packing only.
// ---------------------------------------------------------------------------

typedef __bf16 bf16_t;
typedef __attribute__((ext_vector_type(8)))  __bf16 bf16x8;
typedef __attribute__((ext_vector_type(16))) __bf16 bf16x16;
typedef __attribute__((ext_vector_type(8)))  float  floatx8;

#define BATCH 2
#define SLEN  2048
#define DMODEL 1024
#define NHEAD 16
#define HDIM  64

__device__ __forceinline__ floatx8 wmma_bf16(bf16x16 a, bf16x16 b, floatx8 c) {
    // D(f32 16x16) = A(16x32 bf16) * B(32x16 bf16) + C
    return __builtin_amdgcn_wmma_f32_16x16x32_bf16(
        false, a, false, b, (short)0, c, false, false);
}

__device__ __forceinline__ bf16x16 combine(bf16x8 lo, bf16x8 hi) {
    union { bf16x16 v; bf16x8 h[2]; } u;
    u.h[0] = lo; u.h[1] = hi;
    return u.v;
}

// A-fragment: 16(M) x 32(K) bf16, source row-major [M][K] with stride.
// lanes 0-15: row=lane, K {0..7, 16..23}; lanes 16-31: row=lane-16, K {8..15, 24..31}
__device__ __forceinline__ bf16x16 load_frag_a(const bf16_t* base, int stride, int lane) {
    int r    = lane & 15;
    int koff = (lane & 16) >> 1;            // 0 or 8
    const bf16_t* p = base + (size_t)r * stride + koff;
    return combine(*(const bf16x8*)p, *(const bf16x8*)(p + 16));
}

// B-fragment: 32(K) x 16(N) bf16, source = B^T row-major [N][K] with stride.
// lanes 0-15: col=lane, K {0..7, 16..23}; lanes 16-31: col=lane-16, K {8..15, 24..31}
__device__ __forceinline__ bf16x16 load_frag_b(const bf16_t* base, int stride, int lane) {
    int c    = lane & 15;
    int koff = (lane & 16) >> 1;            // 0 or 8
    const bf16_t* p = base + (size_t)c * stride + koff;
    return combine(*(const bf16x8*)p, *(const bf16x8*)(p + 16));
}

// Async global->LDS 16B move (gfx1250 VGLOBAL, tracked by ASYNCcnt).
__device__ __forceinline__ void async_ld_lds_b128(unsigned lds_addr, const void* gaddr) {
    asm volatile("global_load_async_to_lds_b128 %0, %1, off"
                 :: "v"(lds_addr), "v"((unsigned long long)(uintptr_t)gaddr)
                 : "memory");
}
__device__ __forceinline__ void wait_async0() {
    asm volatile("s_wait_asynccnt 0x0" ::: "memory");
}

// ---------------------------------------------------------------------------
// Conversion / preprocessing kernels
// ---------------------------------------------------------------------------
__global__ void cvt_f32_to_bf16(const float* __restrict__ in,
                                bf16_t* __restrict__ out, int n) {
    int i = blockIdx.x * 256 + threadIdx.x;
    if (i < n) out[i] = (bf16_t)in[i];
}

// Wt[n][k] = W[k][n], 1024x1024, f32 -> bf16
__global__ void transpose_cvt_w(const float* __restrict__ W,
                                bf16_t* __restrict__ Wt) {
    int i = blockIdx.x * 256 + threadIdx.x;
    int row = i >> 10, col = i & 1023;
    Wt[i] = (bf16_t)W[(size_t)col * DMODEL + row];
}

// padb[i] = (1 - mask[i]) * -10000  (additive padding bias, precomputed)
__global__ void build_padb(const float* __restrict__ mask,
                           float* __restrict__ padb, int n) {
    int i = blockIdx.x * 256 + threadIdx.x;
    if (i < n) padb[i] = (1.0f - mask[i]) * -10000.0f;
}

// ---------------------------------------------------------------------------
// WMMA GEMM: C[M=4096][N=1024] = A[M][1024](bf16) * Bt[N][1024]^T + bias
// Block: 256 thr (8 waves), tile 128x128, K-step 64, double-buffered async LDS.
// Wave tile 32x64 -> 16 WMMAs per K-step per wave.
// MODE 0: store bf16 head-major  [(b*16+h)*S + s]*64 + d    (Q, K)
// MODE 1: store bf16 head-transposed [(b*16+h)*64 + d]*S + s (V^T)
// MODE 2: store f32 row-major [m][1024]                      (final out)
// ---------------------------------------------------------------------------
template <int MODE>
__global__ __launch_bounds__(256) void gemm_wmma(const bf16_t* __restrict__ A,
                                                 const bf16_t* __restrict__ Bt,
                                                 const float* __restrict__ bias,
                                                 void* __restrict__ out) {
    // [buf][row*8 + kchunk] covering 128 rows x 64 k; 16 KiB per buffer
    __shared__ bf16x8 sA[2][1024];
    __shared__ bf16x8 sB[2][1024];

    const int tid  = threadIdx.x;
    const int lane = tid & 31;
    const int wave = tid >> 5;
    const int m0 = blockIdx.y * 128;
    const int n0 = blockIdx.x * 128;
    const int wm = (wave >> 1) * 32;     // 0,32,64,96
    const int wn = (wave & 1) * 64;      // 0,64

    floatx8 acc[2][4] = {};

    // Chunk ids this thread stages: c = wave*128 + i*32 + lane (lane-contiguous,
    // so each async instruction fills 512B of consecutive LDS from four 128B global row segments).
    const int cbase = wave * 128 + lane;

    auto issue_tile = [&](int buf, int kk) {
        unsigned la = (unsigned)(uintptr_t)(&sA[buf][0]);
        unsigned lb = (unsigned)(uintptr_t)(&sB[buf][0]);
#pragma unroll
        for (int i = 0; i < 4; ++i) {
            int c   = cbase + i * 32;      // 0..1023 over the block
            int row = c >> 3, kc = c & 7;
            async_ld_lds_b128(la + (unsigned)c * 16u,
                              A  + (size_t)(m0 + row) * DMODEL + kk + kc * 8);
            async_ld_lds_b128(lb + (unsigned)c * 16u,
                              Bt + (size_t)(n0 + row) * DMODEL + kk + kc * 8);
        }
    };

    issue_tile(0, 0);

    const int NSTEP = DMODEL / 64;       // 16
    const int r15  = lane & 15;
    const int hsel = (lane & 16) >> 4;   // half-wave select

    for (int step = 0; step < NSTEP; ++step) {
        wait_async0();                   // my async loads for buf[cur] done
        __syncthreads();                 // everyone's done; prev reads of buf[nxt] done
        if (step + 1 < NSTEP) issue_tile((step + 1) & 1, (step + 1) * 64);

        const bf16x8* cA = &sA[step & 1][0];
        const bf16x8* cB = &sB[step & 1][0];
#pragma unroll
        for (int s = 0; s < 2; ++s) {    // two 32-wide K sub-steps
            const int kc0 = s * 4;
            bf16x16 af[2], bfr[4];
#pragma unroll
            for (int i = 0; i < 2; ++i) {
                int rr = wm + i * 16 + r15;
                af[i] = combine(cA[rr * 8 + kc0 + hsel], cA[rr * 8 + kc0 + hsel + 2]);
            }
#pragma unroll
            for (int j = 0; j < 4; ++j) {
                int rr = wn + j * 16 + r15;
                bfr[j] = combine(cB[rr * 8 + kc0 + hsel], cB[rr * 8 + kc0 + hsel + 2]);
            }
#pragma unroll
            for (int i = 0; i < 2; ++i)
#pragma unroll
                for (int j = 0; j < 4; ++j)
                    acc[i][j] = wmma_bf16(af[i], bfr[j], acc[i][j]);
        }
    }

    // Epilogue (compile-time MODE => branch-free):
    // C layout: lanes 0-15 -> col=lane, row=reg; lanes 16-31 -> row=reg+8
    const int hi8 = (lane & 16) >> 1;
#pragma unroll
    for (int i = 0; i < 2; ++i) {
#pragma unroll
        for (int j = 0; j < 4; ++j) {
            int n = n0 + wn + j * 16 + r15;
            float bv = bias[n];
#pragma unroll
            for (int rr = 0; rr < 8; ++rr) {
                int m = m0 + wm + i * 16 + rr + hi8;
                float val = acc[i][j][rr] + bv;
                if (MODE == 0) {
                    int b = m >> 11, s = m & 2047, hh = n >> 6, d = n & 63;
                    ((bf16_t*)out)[(((size_t)(b * NHEAD + hh)) * SLEN + s) * HDIM + d] = (bf16_t)val;
                } else if (MODE == 1) {
                    int b = m >> 11, s = m & 2047, hh = n >> 6, d = n & 63;
                    ((bf16_t*)out)[(((size_t)(b * NHEAD + hh)) * HDIM + d) * SLEN + s] = (bf16_t)val;
                } else {
                    ((float*)out)[(size_t)m * DMODEL + n] = val;
                }
            }
        }
    }
}

// ---------------------------------------------------------------------------
// Flash attention helpers
// ---------------------------------------------------------------------------
// Online-softmax update on a transposed 16q x 32k score pair; rescales O and
// emits the probability A-fragment (layout-matched: per-lane packing only).
template <bool MASKED>
__device__ __forceinline__ void softmax_pack(floatx8 st[2], float& m_run, float& l_run,
                                             const float* __restrict__ pbk, // padb + b*S + kb
                                             int kb, int q, int hi8,
                                             floatx8 o[4], bf16x16& pfrag) {
    float mx = -1e30f;
#pragma unroll
    for (int t = 0; t < 2; ++t)
#pragma unroll
        for (int r = 0; r < 8; ++r) {
            int kloc = t * 16 + r + hi8;
            float s = st[t][r] * 0.125f + pbk[kloc];            // scale + padding bias
            if (MASKED && (kb + kloc > q)) s += -1e10f;         // causal
            st[t][r] = s;
            mx = fmaxf(mx, s);
        }
    mx = fmaxf(mx, __shfl_xor(mx, 16, 32));                     // other key half
    float m_new = fmaxf(m_run, mx);
    float corr  = __expf(m_run - m_new);
    float rsum  = 0.0f;
#pragma unroll
    for (int t = 0; t < 2; ++t)
#pragma unroll
        for (int r = 0; r < 8; ++r) {
            float p = __expf(st[t][r] - m_new);
            st[t][r] = p;
            rsum += p;
        }
    rsum += __shfl_xor(rsum, 16, 32);
    l_run = l_run * corr + rsum;
    m_run = m_new;

    // Rescale O: factor per output row (row = query) via lane broadcast
    float f[8];
#pragma unroll
    for (int r = 0; r < 8; ++r) f[r] = __shfl(corr, r + hi8, 32);
#pragma unroll
    for (int j = 0; j < 4; ++j)
#pragma unroll
        for (int r = 0; r < 8; ++r) o[j][r] *= f[r];

    // Pack P: St layout == A-fragment layout -> pure per-lane bf16 packing
    union { bf16x16 v; __bf16 e[16]; } pu;
#pragma unroll
    for (int i = 0; i < 8; ++i) {
        pu.e[i]     = (__bf16)st[0][i];   // keys {0..7}/{8..15}
        pu.e[8 + i] = (__bf16)st[1][i];   // keys {16..23}/{24..31}
    }
    pfrag = pu.v;
}

// ---------------------------------------------------------------------------
// Flash attention, one wave per (b, h, 32-query pair of 16-row tiles).
// K/V fragments are loaded once per key tile and feed BOTH query sub-tiles:
// 16 WMMAs per 32-key tile against 8 fragment loads.
// Exactly one causal-masked tail tile (kb == qbase0) since qbase0 is 32-aligned.
// Q,K: [B][H][S][64] bf16 ; Vt: [B][H][64][S] bf16 ; ctx out: [B][S][1024] bf16
// ---------------------------------------------------------------------------
__global__ __launch_bounds__(128) void attn_flash(const bf16_t* __restrict__ Q,
                                                  const bf16_t* __restrict__ K,
                                                  const bf16_t* __restrict__ Vt,
                                                  const float* __restrict__ padb,
                                                  bf16_t* __restrict__ ctx) {
    const int lane = threadIdx.x & 31;
    const int gw   = blockIdx.x * 4 + (threadIdx.x >> 5);
    const int qpair = gw & (SLEN / 32 - 1);      // 0..63
    const int bh    = gw >> 6;                   // 0..31
    const int h = bh & (NHEAD - 1), b = bh >> 4;
    const int qbase0 = qpair * 32;               // 32-aligned
    const int qbase1 = qbase0 + 16;

    const bf16_t* Qh = Q  + ((size_t)(b * NHEAD + h)) * SLEN * HDIM;
    const bf16_t* Kh = K  + ((size_t)(b * NHEAD + h)) * SLEN * HDIM;
    const bf16_t* Vh = Vt + ((size_t)(b * NHEAD + h)) * HDIM * SLEN;
    const float*  pb = padb + (size_t)b * SLEN;

    // Q^T B-fragments for both sub-tiles (hd chunks 0-31, 32-63); lane-col = query
    bf16x16 qf[2][2];
    qf[0][0] = load_frag_b(Qh + (size_t)qbase0 * HDIM + 0,  HDIM, lane);
    qf[0][1] = load_frag_b(Qh + (size_t)qbase0 * HDIM + 32, HDIM, lane);
    qf[1][0] = load_frag_b(Qh + (size_t)qbase1 * HDIM + 0,  HDIM, lane);
    qf[1][1] = load_frag_b(Qh + (size_t)qbase1 * HDIM + 32, HDIM, lane);

    floatx8 o0[4] = {}, o1[4] = {};   // ctx accumulators: 2 x (16 q x 64 d)
    float m0 = -1e30f, l0 = 0.0f, m1 = -1e30f, l1 = 0.0f;

    const int q0  = qbase0 + (lane & 15);
    const int q1  = qbase1 + (lane & 15);
    const int hi8 = (lane & 16) >> 1;

    auto key_tile = [&](int kb, auto masked) {
        // Shared K fragments: [16-key sub-tile][hd chunk]
        bf16x16 kf[2][2];
#pragma unroll
        for (int t = 0; t < 2; ++t) {
            const bf16_t* kp = Kh + (size_t)(kb + t * 16) * HDIM;
            kf[t][0] = load_frag_a(kp + 0,  HDIM, lane);
            kf[t][1] = load_frag_a(kp + 32, HDIM, lane);
        }
        // scores^T for both query sub-tiles (K frags reused)
        floatx8 s0[2], s1[2];
#pragma unroll
        for (int t = 0; t < 2; ++t) {
            floatx8 z = {};
            s0[t] = wmma_bf16(kf[t][1], qf[0][1], wmma_bf16(kf[t][0], qf[0][0], z));
            s1[t] = wmma_bf16(kf[t][1], qf[1][1], wmma_bf16(kf[t][0], qf[1][0], z));
        }
        bf16x16 p0, p1;
        softmax_pack<masked.value>(s0, m0, l0, pb + kb, kb, q0, hi8, o0, p0);
        softmax_pack<masked.value>(s1, m1, l1, pb + kb, kb, q1, hi8, o1, p1);
        // O += P @ V : V fragments shared by both sub-tiles
#pragma unroll
        for (int j = 0; j < 4; ++j) {
            bf16x16 vf = load_frag_b(Vh + (size_t)(j * 16) * SLEN + kb, SLEN, lane);
            o0[j] = wmma_bf16(p0, vf, o0[j]);
            o1[j] = wmma_bf16(p1, vf, o1[j]);
        }
    };

    for (int kb = 0; kb < qbase0; kb += 32)
        key_tile(kb, std::integral_constant<bool, false>{});
    key_tile(qbase0, std::integral_constant<bool, true>{});

    // --- finalize: divide by row-sum, store ctx [b][s][h*64+d] bf16 ---
    float li0 = 1.0f / l0, li1 = 1.0f / l1;
    float f0[8], f1[8];
#pragma unroll
    for (int r = 0; r < 8; ++r) {
        f0[r] = __shfl(li0, r + hi8, 32);
        f1[r] = __shfl(li1, r + hi8, 32);
    }
#pragma unroll
    for (int j = 0; j < 4; ++j)
#pragma unroll
        for (int r = 0; r < 8; ++r) {
            int dcol = h * HDIM + j * 16 + (lane & 15);
            int s0r = qbase0 + r + hi8;
            int s1r = qbase1 + r + hi8;
            ctx[((size_t)(b * SLEN + s0r)) * DMODEL + dcol] = (bf16_t)(o0[j][r] * f0[r]);
            ctx[((size_t)(b * SLEN + s1r)) * DMODEL + dcol] = (bf16_t)(o1[j][r] * f1[r]);
        }
}

// ---------------------------------------------------------------------------
// Host launcher
// ---------------------------------------------------------------------------
extern "C" void kernel_launch(void* const* d_in, const int* in_sizes, int n_in,
                              void* d_out, int out_size, void* d_ws, size_t ws_size,
                              hipStream_t stream) {
    const float* hidden = (const float*)d_in[0];
    const float* amask  = (const float*)d_in[1];
    const float* Wq = (const float*)d_in[2];  const float* bq = (const float*)d_in[3];
    const float* Wk = (const float*)d_in[4];  const float* bk = (const float*)d_in[5];
    const float* Wv = (const float*)d_in[6];  const float* bv = (const float*)d_in[7];
    const float* Wo = (const float*)d_in[8];  const float* bo = (const float*)d_in[9];
    float* out = (float*)d_out;

    char* ws = (char*)d_ws;
    const size_t MiB = 1u << 20;
    bf16_t* Xbf = (bf16_t*)(ws + 0);          //  8 MiB [4096][1024]
    bf16_t* WqT = (bf16_t*)(ws +  8 * MiB);   //  2 MiB each
    bf16_t* WkT = (bf16_t*)(ws + 10 * MiB);
    bf16_t* WvT = (bf16_t*)(ws + 12 * MiB);
    bf16_t* WoT = (bf16_t*)(ws + 14 * MiB);
    bf16_t* Qb  = (bf16_t*)(ws + 16 * MiB);   //  8 MiB [B][H][S][64]
    bf16_t* Kb  = (bf16_t*)(ws + 24 * MiB);   //  8 MiB
    bf16_t* Vtb = (bf16_t*)(ws + 32 * MiB);   //  8 MiB [B][H][64][S]
    bf16_t* Ctx = (bf16_t*)(ws + 40 * MiB);   //  8 MiB [B][S][1024]
    float*  Pad = (float*)(ws + 48 * MiB);    // 16 KiB [B][S]

    const int NX = BATCH * SLEN * DMODEL;     // 4M
    cvt_f32_to_bf16<<<NX / 256, 256, 0, stream>>>(hidden, Xbf, NX);
    const int NW = DMODEL * DMODEL;           // 1M
    transpose_cvt_w<<<NW / 256, 256, 0, stream>>>(Wq, WqT);
    transpose_cvt_w<<<NW / 256, 256, 0, stream>>>(Wk, WkT);
    transpose_cvt_w<<<NW / 256, 256, 0, stream>>>(Wv, WvT);
    transpose_cvt_w<<<NW / 256, 256, 0, stream>>>(Wo, WoT);
    build_padb<<<(BATCH * SLEN) / 256, 256, 0, stream>>>(amask, Pad, BATCH * SLEN);

    dim3 ggrid(DMODEL / 128, (BATCH * SLEN) / 128);   // (8, 32)
    gemm_wmma<0><<<ggrid, 256, 0, stream>>>(Xbf, WqT, bq, (void*)Qb);
    gemm_wmma<0><<<ggrid, 256, 0, stream>>>(Xbf, WkT, bk, (void*)Kb);
    gemm_wmma<1><<<ggrid, 256, 0, stream>>>(Xbf, WvT, bv, (void*)Vtb);

    const int nwaves = BATCH * NHEAD * (SLEN / 32);   // 2048
    attn_flash<<<nwaves / 4, 128, 0, stream>>>(Qb, Kb, Vtb, Pad, Ctx);

    gemm_wmma<2><<<ggrid, 256, 0, stream>>>(Ctx, WoT, bo, (void*)out);
}
